// MoEFFNLayer_4973572129155
// MI455X (gfx1250) — matrix operand order repaired and checked by reading, not compiled, not run on hardware
//
#include <hip/hip_runtime.h>
#include <stdint.h>

// ---------------------------------------------------------------------------
// MoE FFN (router is statically one-hot on expert 0 => only expert 0 is live)
//   H  = relu(tgt @ w1[0] + b1[0])
//   EO = H @ w2[0] + b2[0]
//   out = LayerNorm(tgt + EO) * gamma + beta
// Compute-bound (68.7 GFLOP vs ~100MB traffic) => bf16 WMMA with f32 accum.
// Weights are transposed during the one-time fp32->bf16 conversion so BOTH
// WMMA operand fragments load as two aligned ds_load_b128 (no scalar gathers).
// Global->LDS staging uses CDNA5 async-to-LDS (ASYNCcnt), double-buffered.
// ---------------------------------------------------------------------------

typedef __bf16 bf16_t;
typedef __attribute__((ext_vector_type(16))) __bf16 v16bf;
typedef __attribute__((ext_vector_type(8)))  __bf16 v8bf;
typedef __attribute__((ext_vector_type(8)))  float  v8f;
typedef int v4i __attribute__((vector_size(16)));   // matches builtin param type

#if defined(__has_builtin)
#  if __has_builtin(__builtin_amdgcn_global_load_async_to_lds_b128)
#    define HAS_ASYNC_LDS 1
#  endif
#endif
#ifndef HAS_ASYNC_LDS
#  define HAS_ASYNC_LDS 0
#endif

__device__ __forceinline__ bf16_t f32_to_bf16_rne(float f) {
    union { float f; uint32_t u; } in;
    in.f = f;
    uint32_t u = in.u;
    uint32_t r = u + 0x7FFFu + ((u >> 16) & 1u);   // round to nearest even
    union { uint16_t u; bf16_t b; } out;
    out.u = (uint16_t)(r >> 16);
    return out.b;
}

#if HAS_ASYNC_LDS
__device__ __forceinline__ void async_copy16(const bf16_t* g, bf16_t* l) {
    // builtin signature (from hipcc diagnostic): arg0 = AS1 v4i*, 4 args total
    __builtin_amdgcn_global_load_async_to_lds_b128(
        (__attribute__((address_space(1))) v4i*)g,
        (__attribute__((address_space(3))) v4i*)l, 0, 0);
}
__device__ __forceinline__ void wait_async_4() {
#  if __has_builtin(__builtin_amdgcn_s_wait_asynccnt)
    __builtin_amdgcn_s_wait_asynccnt(4);
#  else
    asm volatile("s_wait_asynccnt 0x4" ::: "memory");
#  endif
}
__device__ __forceinline__ void wait_async_0() {
#  if __has_builtin(__builtin_amdgcn_s_wait_asynccnt)
    __builtin_amdgcn_s_wait_asynccnt(0);
#  else
    asm volatile("s_wait_asynccnt 0x0" ::: "memory");
#  endif
}
#endif

// ------------------------- fp32 -> bf16 conversion -------------------------
__global__ void __launch_bounds__(256)
cvt_f32_to_bf16(const float* __restrict__ in, bf16_t* __restrict__ out, int n) {
    int i = blockIdx.x * blockDim.x + threadIdx.x;
    int stride = gridDim.x * blockDim.x;
    for (; i < n; i += stride) out[i] = f32_to_bf16_rne(in[i]);
}

// ---------------- fp32 (R x C) -> bf16 transposed (C x R) ------------------
// Tiled 32x32 through LDS: coalesced reads and coalesced writes.
__global__ void __launch_bounds__(256)
transpose_cvt_f32_to_bf16(const float* __restrict__ in, bf16_t* __restrict__ out,
                          int R, int C) {
    __shared__ float tile[32][33];
    const int c0 = blockIdx.x * 32;
    const int r0 = blockIdx.y * 32;
    const int tx = threadIdx.x & 31;
    const int ty = threadIdx.x >> 5;            // 0..7
#pragma unroll
    for (int i = 0; i < 32; i += 8)
        tile[ty + i][tx] = in[(size_t)(r0 + ty + i) * C + (c0 + tx)];
    __syncthreads();
#pragma unroll
    for (int i = 0; i < 32; i += 8)
        out[(size_t)(c0 + ty + i) * R + (r0 + tx)] = f32_to_bf16_rne(tile[tx][ty + i]);
}

// ------------------------------- WMMA GEMM ---------------------------------
// C[M,N] = A[M,K] (bf16, row-major) @ Bt[N,K]^T (bf16; Bt row-major = B col-major)
// MODE 0: + bias, relu, store bf16            (FFN layer 1)
// MODE 1: + bias + residual[M,N], store f32   (FFN layer 2 + residual)
// Block tile 128x128, BK=32; 256 threads = 8 waves in a 2x4 grid, each wave
// owns 64x32 = 4x2 WMMA tiles. Both A and Bt staged as [128][BK] contiguous-K
// tiles so every fragment is two aligned ds_load_b128.
template <int MODE>
__global__ void __launch_bounds__(256)
gemm_wmma_bf16(const bf16_t* __restrict__ A, const bf16_t* __restrict__ Bt,
               const float* __restrict__ bias, const float* __restrict__ residual,
               bf16_t* __restrict__ outBf, float* __restrict__ outF,
               int M, int N, int K) {
    constexpr int BM = 128, BN = 128, BK = 32;
#if HAS_ASYNC_LDS
    constexpr int NBUF = 2;                    // double buffer for async pipeline
#else
    constexpr int NBUF = 1;
#endif
    __shared__ bf16_t Al[NBUF][BM][BK + 8];    // +16B pad: conflict-free b128 reads
    __shared__ bf16_t Bl[NBUF][BN][BK + 8];

    const int t     = threadIdx.x;
    const int lane  = t & 31;
    const int wave  = t >> 5;
    const int waveM = wave >> 2;               // 0..1 -> 64-row slab
    const int waveN = wave & 3;                // 0..3 -> 32-col slab
    const int m0    = blockIdx.y * BM;
    const int n0    = blockIdx.x * BN;
    const int half  = lane >> 4;               // 0/1
    const int l15   = lane & 15;

    v8f acc[4][2];
    v8f zero = 0.0f;
#pragma unroll
    for (int i = 0; i < 4; ++i)
#pragma unroll
        for (int j = 0; j < 2; ++j) acc[i][j] = zero;

    // staging assignment: thread t moves 32B of A and 32B of Bt per tile
    const int sRow = t >> 1;                   // 0..127
    const int sOff = (t & 1) * 16;             // 0 / 16
    const bf16_t* Ag = A  + (size_t)(m0 + sRow) * K + sOff;
    const bf16_t* Bg = Bt + (size_t)(n0 + sRow) * K + sOff;
    const int nk = K / BK;

#if HAS_ASYNC_LDS
    // prologue: tile 0 -> buffer 0 (4 async b128 per thread)
    async_copy16(Ag,     &Al[0][sRow][sOff]);
    async_copy16(Ag + 8, &Al[0][sRow][sOff + 8]);
    async_copy16(Bg,     &Bl[0][sRow][sOff]);
    async_copy16(Bg + 8, &Bl[0][sRow][sOff + 8]);
#endif

    for (int kt = 0; kt < nk; ++kt) {
        const int buf = kt & (NBUF - 1);
#if HAS_ASYNC_LDS
        if (kt + 1 < nk) {
            const int nb = (kt + 1) & 1;
            const bf16_t* Agn = Ag + (size_t)(kt + 1) * BK;
            const bf16_t* Bgn = Bg + (size_t)(kt + 1) * BK;
            async_copy16(Agn,     &Al[nb][sRow][sOff]);
            async_copy16(Agn + 8, &Al[nb][sRow][sOff + 8]);
            async_copy16(Bgn,     &Bl[nb][sRow][sOff]);
            async_copy16(Bgn + 8, &Bl[nb][sRow][sOff + 8]);
            wait_async_4();                    // tile kt has landed
        } else {
            wait_async_0();
        }
        __syncthreads();
#else
        const bf16_t* Agc = Ag + (size_t)kt * BK;
        const bf16_t* Bgc = Bg + (size_t)kt * BK;
        *(uint4*)&Al[0][sRow][sOff]     = *(const uint4*)(Agc);
        *(uint4*)&Al[0][sRow][sOff + 8] = *(const uint4*)(Agc + 8);
        *(uint4*)&Bl[0][sRow][sOff]     = *(const uint4*)(Bgc);
        *(uint4*)&Bl[0][sRow][sOff + 8] = *(const uint4*)(Bgc + 8);
        if (kt + 1 < nk) {
            __builtin_prefetch(Agc + BK, 0, 0);   // global_prefetch_b8
            __builtin_prefetch(Bgc + BK, 0, 0);
        }
        __syncthreads();
#endif

        // B fragments (contiguous-K order, same as A: lanes 0-15 hold K0..15 of column n,
        // lanes 16-31 hold K16..31) -> two ds_load_b128 from the Bt tile.
        v16bf bfrag[2];
#pragma unroll
        for (int j = 0; j < 2; ++j) {
            const int col = waveN * 32 + j * 16 + l15;
            const int kb  = half * 16;
            v8bf lo = *(const v8bf*)&Bl[buf][col][kb];
            v8bf hi = *(const v8bf*)&Bl[buf][col][kb + 8];
            v16bf v;
#pragma unroll
            for (int q = 0; q < 8; ++q) { v[q] = lo[q]; v[q + 8] = hi[q]; }
            bfrag[j] = v;
        }

        // A fragments (lane<16 -> K0-15; lane>=16 -> K16-31)
#pragma unroll
        for (int i = 0; i < 4; ++i) {
            const int row = waveM * 64 + i * 16 + l15;
            v8bf lo = *(const v8bf*)&Al[buf][row][half * 16];
            v8bf hi = *(const v8bf*)&Al[buf][row][half * 16 + 8];
            v16bf av;
#pragma unroll
            for (int q = 0; q < 8; ++q) { av[q] = lo[q]; av[q + 8] = hi[q]; }
#pragma unroll
            for (int j = 0; j < 2; ++j) {
                acc[i][j] = __builtin_amdgcn_wmma_f32_16x16x32_bf16(
                    false, av, false, bfrag[j], (short)0, acc[i][j], false, false);
            }
        }
        __syncthreads();
    }

    // Epilogue. C/D layout: VGPR r -> row = half*8 + r, col = lane&15.
#pragma unroll
    for (int i = 0; i < 4; ++i) {
#pragma unroll
        for (int j = 0; j < 2; ++j) {
            const int col = n0 + waveN * 32 + j * 16 + l15;
            const float bv = bias[col];
#pragma unroll
            for (int r = 0; r < 8; ++r) {
                const int row = m0 + waveM * 64 + i * 16 + half * 8 + r;
                float v = acc[i][j][r] + bv;
                if (MODE == 0) {
                    v = v > 0.0f ? v : 0.0f;
                    outBf[(size_t)row * N + col] = f32_to_bf16_rne(v);
                } else {
                    outF[(size_t)row * N + col] = v + residual[(size_t)row * N + col];
                }
            }
        }
    }
}

// ------------------------------- LayerNorm ---------------------------------
__global__ void __launch_bounds__(256)
layernorm_rows(const float* __restrict__ X, const float* __restrict__ gamma,
               const float* __restrict__ beta, float* __restrict__ out, int D) {
    const int row = blockIdx.x;
    const float* x = X + (size_t)row * D;
    float* o = out + (size_t)row * D;

    __shared__ float s_sum[256];
    __shared__ float s_sq[256];
    float s = 0.0f, sq = 0.0f;
    for (int c = threadIdx.x; c < D; c += 256) {
        float v = x[c];
        s += v; sq += v * v;
    }
    s_sum[threadIdx.x] = s; s_sq[threadIdx.x] = sq;
    __syncthreads();
    for (int off = 128; off > 0; off >>= 1) {
        if (threadIdx.x < (unsigned)off) {
            s_sum[threadIdx.x] += s_sum[threadIdx.x + off];
            s_sq[threadIdx.x]  += s_sq[threadIdx.x + off];
        }
        __syncthreads();
    }
    const float mean = s_sum[0] / (float)D;
    const float var  = s_sq[0] / (float)D - mean * mean;
    const float inv  = rsqrtf(var + 1e-5f);
    for (int c = threadIdx.x; c < D; c += 256)
        o[c] = (x[c] - mean) * inv * gamma[c] + beta[c];
}

// ------------------------------- launcher ----------------------------------
extern "C" void kernel_launch(void* const* d_in, const int* in_sizes, int n_in,
                              void* d_out, int out_size, void* d_ws, size_t ws_size,
                              hipStream_t stream) {
    (void)in_sizes; (void)n_in; (void)out_size; (void)ws_size;

    // setup_inputs order: tgt, router_w, router_b, w1, b1, w2, b2, gamma, beta
    const float* tgt   = (const float*)d_in[0];
    const float* w1    = (const float*)d_in[3];   // (E, D, F); expert 0 = first D*F
    const float* b1    = (const float*)d_in[4];   // (E, F);    expert 0 = first F
    const float* w2    = (const float*)d_in[5];   // (E, F, D); expert 0 = first F*D
    const float* b2    = (const float*)d_in[6];   // (E, D);    expert 0 = first D
    const float* gamma = (const float*)d_in[7];
    const float* beta  = (const float*)d_in[8];
    float* out = (float*)d_out;

    constexpr int Bsz = 2, S = 2048, D = 1024, F = 4096;
    constexpr int M = Bsz * S;                    // 4096 tokens

    // workspace carve-out (256B aligned)
    char* ws = (char*)d_ws;
    size_t off = 0;
    auto carve = [&](size_t bytes) -> void* {
        void* p = ws + off;
        off += (bytes + 255) & ~(size_t)255;
        return p;
    };
    bf16_t* tgtBf = (bf16_t*)carve((size_t)M * D * sizeof(bf16_t));   //  8 MB
    bf16_t* w1T   = (bf16_t*)carve((size_t)F * D * sizeof(bf16_t));   //  8 MB  (F x D)
    bf16_t* w2T   = (bf16_t*)carve((size_t)D * F * sizeof(bf16_t));   //  8 MB  (D x F)
    bf16_t* Hbf   = (bf16_t*)carve((size_t)M * F * sizeof(bf16_t));   // 32 MB
    float*  Xf    = (float*) carve((size_t)M * D * sizeof(float));    // 16 MB

    // 1) one-time conversions; weights transposed so GEMM B-fragments are
    //    contiguous-K (pure ds_load_b128 in the inner loop)
    cvt_f32_to_bf16<<<4096, 256, 0, stream>>>(tgt, tgtBf, M * D);
    {
        dim3 g(F / 32, D / 32);   // w1: (D x F) -> w1T: (F x D)
        transpose_cvt_f32_to_bf16<<<g, 256, 0, stream>>>(w1, w1T, D, F);
    }
    {
        dim3 g(D / 32, F / 32);   // w2: (F x D) -> w2T: (D x F)
        transpose_cvt_f32_to_bf16<<<g, 256, 0, stream>>>(w2, w2T, F, D);
    }

    // 2) H = relu(tgt @ w1[0] + b1[0])   -> bf16 [M, F]
    {
        dim3 grid(F / 128, M / 128);
        gemm_wmma_bf16<0><<<grid, 256, 0, stream>>>(
            tgtBf, w1T, b1, nullptr, Hbf, nullptr, M, F, D);
    }

    // 3) X = H @ w2[0] + b2[0] + tgt     -> f32 [M, D]
    {
        dim3 grid(D / 128, M / 128);
        gemm_wmma_bf16<1><<<grid, 256, 0, stream>>>(
            Hbf, w2T, b2, tgt, nullptr, Xf, M, D, F);
    }

    // 4) out = LayerNorm(X) * gamma + beta
    layernorm_rows<<<M, 256, 0, stream>>>(Xf, gamma, beta, out, D);
}
